// Transformer_27676769255494
// MI455X (gfx1250) — matrix operand order repaired and checked by reading, not compiled, not run on hardware
//
#include <hip/hip_runtime.h>

// ---------------------------------------------------------------------------
// MI455X (gfx1250, wave32) Transformer forward pass.
// All GEMMs run on the matrix cores via V_WMMA_F32_16X16X32_BF16:
// fp32 inputs are converted (RNE) to bf16 while staging tiles in LDS, and
// accumulation stays fp32 — the precision/throughput sweet spot for this
// fp32-weight model on CDNA5. Compute is GEMM-bound (~300 GFLOP vs ~25us of
// HBM traffic at 23.3 TB/s), so WMMA coverage of every matmul is the win.
// ---------------------------------------------------------------------------

typedef __attribute__((ext_vector_type(16))) __bf16 v16bf;
typedef __attribute__((ext_vector_type(8)))  float  v8f;

struct alignas(16) I4 { unsigned int a, b, c, d; };

__device__ __forceinline__ unsigned short f32_to_bf16(float f) {
  union { float f; unsigned u; } x; x.f = f;
  unsigned u = x.u + 0x7FFFu + ((x.u >> 16) & 1u);   // round-to-nearest-even
  return (unsigned short)(u >> 16);
}

__device__ __forceinline__ void store4_bf16(unsigned short* d, float4 f) {
  unsigned int* p = reinterpret_cast<unsigned int*>(d);   // 4B aligned by construction
  p[0] = ((unsigned)f32_to_bf16(f.y) << 16) | f32_to_bf16(f.x);
  p[1] = ((unsigned)f32_to_bf16(f.w) << 16) | f32_to_bf16(f.z);
}

// Load a 32-byte WMMA fragment from LDS given the two 16-byte chunks.
__device__ __forceinline__ v16bf load_frag2(const unsigned short* base, int second16B) {
  union { I4 i[2]; v16bf v; } u;
  const I4* p = reinterpret_cast<const I4*>(base);
  u.i[0] = p[0];
  u.i[1] = p[second16B];   // A frag: +32B (skip 16 elems); B frag: +32B (skip 16 elems)
  return u.v;
}

// ---------------------------------------------------------------------------
// GEMM: C[M,N] = A[M,K] @ B + (bias) (+ReLU). TRANSB: B given as [N,K]
// row-major (so its LDS N-major staging is a straight copy — used for q@k^T);
// otherwise B is [K,N] and is transposed while converting into LDS.
// Block tile 128x64, BK=32, 256 threads = 8 waves in a 4x2 grid; each wave
// owns a 32x32 quad of 16x16x32 bf16 WMMA tiles (4 x v8f accumulators).
// All shapes used are exact multiples of the tiles (M=2048; N,K per above).
// ---------------------------------------------------------------------------
#define LDSTR 40   // padded bf16 row stride (80B = 5*16B: keeps 16B alignment, kills bank conflicts)

template <bool TRANSB, bool BIAS, bool RELU>
__global__ __launch_bounds__(256) void gemm_wmma(
    const float* __restrict__ A, const float* __restrict__ B,
    const float* __restrict__ bias, float* __restrict__ C,
    int M, int N, int K, int lda, int ldb, int ldc) {
  __shared__ unsigned short As[128 * LDSTR];
  __shared__ unsigned short Bs[64 * LDSTR];

  const int tid   = threadIdx.x;
  const int lane  = tid & 31;
  const int wave  = tid >> 5;
  const int waveM = wave >> 1;       // 0..3
  const int waveN = wave & 1;        // 0..1
  const int mBase = blockIdx.y * 128;
  const int nBase = blockIdx.x * 64;
  const int lr = lane & 15;          // row/col within 16x16 tile
  const int hs = lane >> 4;          // half-wave select
  const int kbA = hs * 8;            // A-frag K sub-block (ISA 16-bit A layout)

  v8f acc[2][2] = {};

  for (int k0 = 0; k0 < K; k0 += 32) {
    // ---- stage A tile (128x32 f32 -> bf16, row-major K-contiguous) ----
    {
      const int r0 = tid >> 3, c0 = (tid & 7) * 4;
      #pragma unroll
      for (int rr = 0; rr < 4; ++rr) {
        const int row = r0 + rr * 32;
        float4 f = *reinterpret_cast<const float4*>(
            &A[(size_t)(mBase + row) * lda + k0 + c0]);
        store4_bf16(&As[row * LDSTR + c0], f);
      }
    }
    // ---- stage B tile, N-major ([n][k] in LDS) ----
    if (TRANSB) {  // B is [N,K]: direct copy
      const int n0 = tid >> 3, c0 = (tid & 7) * 4;
      #pragma unroll
      for (int nn = 0; nn < 2; ++nn) {
        const int n = n0 + nn * 32;
        float4 f = *reinterpret_cast<const float4*>(
            &B[(size_t)(nBase + n) * ldb + k0 + c0]);
        store4_bf16(&Bs[n * LDSTR + c0], f);
      }
    } else {       // B is [K,N]: transpose while staging
      const int kr0 = tid >> 4, n0 = (tid & 15) * 4;
      #pragma unroll
      for (int kk = 0; kk < 2; ++kk) {
        const int kr = kr0 + kk * 16;
        float4 f = *reinterpret_cast<const float4*>(
            &B[(size_t)(k0 + kr) * ldb + nBase + n0]);
        Bs[(n0 + 0) * LDSTR + kr] = f32_to_bf16(f.x);
        Bs[(n0 + 1) * LDSTR + kr] = f32_to_bf16(f.y);
        Bs[(n0 + 2) * LDSTR + kr] = f32_to_bf16(f.z);
        Bs[(n0 + 3) * LDSTR + kr] = f32_to_bf16(f.w);
      }
    }
    __syncthreads();

    // ---- WMMA: 2x2 16x16 tiles per wave ----
    v16bf afrag[2], bfrag[2];
    #pragma unroll
    for (int mt = 0; mt < 2; ++mt) {
      const int row = waveM * 32 + mt * 16 + lr;
      // A 16x32 bf16 layout: lane holds K {kbA..kbA+7} and {16+kbA..16+kbA+7}
      afrag[mt] = load_frag2(&As[row * LDSTR + kbA], 2);
    }
    #pragma unroll
    for (int nt = 0; nt < 2; ++nt) {
      const int n = waveN * 32 + nt * 16 + lr;
      // B 32x16 bf16 layout: lane holds K {hs*8..hs*8+7} and {16+hs*8..16+hs*8+7}
      bfrag[nt] = load_frag2(&Bs[n * LDSTR + hs * 8], 2);
    }
    #pragma unroll
    for (int mt = 0; mt < 2; ++mt)
      #pragma unroll
      for (int nt = 0; nt < 2; ++nt)
        acc[mt][nt] = __builtin_amdgcn_wmma_f32_16x16x32_bf16(
            false, afrag[mt], false, bfrag[nt], (short)0, acc[mt][nt],
            false, false);
    __syncthreads();
  }

  // ---- epilogue: C/D layout — VGPR j holds M = hs*8 + j, N = lane&15 ----
  #pragma unroll
  for (int mt = 0; mt < 2; ++mt)
    #pragma unroll
    for (int nt = 0; nt < 2; ++nt) {
      const int gn = nBase + waveN * 32 + nt * 16 + lr;
      const float bv = BIAS ? bias[gn] : 0.0f;
      #pragma unroll
      for (int j = 0; j < 8; ++j) {
        const int gm = mBase + waveM * 32 + mt * 16 + hs * 8 + j;
        float v = acc[mt][nt][j] + bv;
        if (RELU) v = fmaxf(v, 0.0f);
        C[(size_t)gm * ldc + gn] = v;
      }
    }
}

// ---------------------------------------------------------------------------
// Embedding gather + sinusoidal positional encoding (faithful: angle uses the
// raw dim index i, even i -> sin, odd i -> cos).
// ---------------------------------------------------------------------------
__global__ __launch_bounds__(256) void embed_pos(
    const int* __restrict__ tok, const float* __restrict__ emb,
    float* __restrict__ out) {
  const int idx = blockIdx.x * 256 + threadIdx.x;   // 2048*512
  const int t = idx >> 9, d = idx & 511;
  // 10000^(-2d/512) = exp(-(2d/512)*ln 10000)
  const float ang = (float)t * __expf((-2.0f / 512.0f) * (float)d * 9.210340371976184f);
  const float pe = (d & 1) ? __cosf(ang) : __sinf(ang);
  out[idx] = emb[(size_t)tok[t] * 512 + d] + pe;
}

// ---------------------------------------------------------------------------
// Attention softmax over rows of [2048,2048] with the reference's quirks:
// tril sets masked entries to 0 (NOT -inf), then everything is scaled by
// 1/sqrt(d_model)=1/sqrt(512), then ordinary softmax over all 2048 entries.
// ---------------------------------------------------------------------------
__global__ __launch_bounds__(256) void attn_softmax(float* __restrict__ s, int causal) {
  __shared__ float red[256];
  const int r = blockIdx.x, t = threadIdx.x;
  const float scale = 0.044194173824159216f;  // 1/sqrt(512)
  const size_t base = (size_t)r * 2048;
  float vals[8];
  float mx = -3.4e38f;
  #pragma unroll
  for (int i = 0; i < 8; ++i) {
    const int c = t + i * 256;
    float v = s[base + c];
    if (causal && c > r) v = 0.0f;
    v *= scale;
    vals[i] = v;
    mx = fmaxf(mx, v);
  }
  red[t] = mx; __syncthreads();
  for (int o = 128; o > 0; o >>= 1) { if (t < o) red[t] = fmaxf(red[t], red[t + o]); __syncthreads(); }
  mx = red[0]; __syncthreads();
  float sum = 0.0f;
  #pragma unroll
  for (int i = 0; i < 8; ++i) { vals[i] = __expf(vals[i] - mx); sum += vals[i]; }
  red[t] = sum; __syncthreads();
  for (int o = 128; o > 0; o >>= 1) { if (t < o) red[t] += red[t + o]; __syncthreads(); }
  const float inv = 1.0f / red[0];
  #pragma unroll
  for (int i = 0; i < 8; ++i) s[base + t + i * 256] = vals[i] * inv;
}

// ---------------------------------------------------------------------------
// Residual add + LayerNorm over D=512, in place on x.
// ---------------------------------------------------------------------------
__global__ __launch_bounds__(256) void ln_res(
    float* __restrict__ x, const float* __restrict__ sub,
    const float* __restrict__ g, const float* __restrict__ b) {
  __shared__ float red[256];
  const int r = blockIdx.x, t = threadIdx.x;
  const size_t base = (size_t)r * 512;
  const float y0 = x[base + t] + sub[base + t];
  const float y1 = x[base + t + 256] + sub[base + t + 256];
  red[t] = y0 + y1; __syncthreads();
  for (int o = 128; o > 0; o >>= 1) { if (t < o) red[t] += red[t + o]; __syncthreads(); }
  const float mean = red[0] * (1.0f / 512.0f);
  __syncthreads();
  const float d0 = y0 - mean, d1 = y1 - mean;
  red[t] = d0 * d0 + d1 * d1; __syncthreads();
  for (int o = 128; o > 0; o >>= 1) { if (t < o) red[t] += red[t + o]; __syncthreads(); }
  const float inv = rsqrtf(red[0] * (1.0f / 512.0f) + 1e-5f);
  x[base + t]       = d0 * inv * g[t]       + b[t];
  x[base + t + 256] = d1 * inv * g[t + 256] + b[t + 256];
}

// ---------------------------------------------------------------------------
// Head-concat fold: tile(head,NH) @ Wout == head @ sum_h Wout[h*64+kd, :].
// Wsum[64,512] precomputed here; turns the Wout GEMM into K=64.
// ---------------------------------------------------------------------------
__global__ __launch_bounds__(256) void wout_fold(
    const float* __restrict__ w, float* __restrict__ wsum) {
  const int idx = blockIdx.x * 256 + threadIdx.x;   // 64*512
  const int kd = idx >> 9, n = idx & 511;
  float s = 0.0f;
  #pragma unroll
  for (int h = 0; h < 8; ++h) s += w[(size_t)(h * 64 + kd) * 512 + n];
  wsum[idx] = s;
}

// ---------------------------------------------------------------------------
// Row softmax over V=32000 (axis=1), in place on the logits/output buffer.
// ---------------------------------------------------------------------------
__global__ __launch_bounds__(256) void vocab_softmax(float* __restrict__ lg) {
  __shared__ float red[256];
  const int r = blockIdx.x, t = threadIdx.x;
  const size_t base = (size_t)r * 32000;
  float mx = -3.4e38f;
  for (int c = t; c < 32000; c += 256) mx = fmaxf(mx, lg[base + c]);
  red[t] = mx; __syncthreads();
  for (int o = 128; o > 0; o >>= 1) { if (t < o) red[t] = fmaxf(red[t], red[t + o]); __syncthreads(); }
  mx = red[0]; __syncthreads();
  float sum = 0.0f;
  for (int c = t; c < 32000; c += 256) {
    const float e = __expf(lg[base + c] - mx);
    lg[base + c] = e;
    sum += e;
  }
  red[t] = sum; __syncthreads();
  for (int o = 128; o > 0; o >>= 1) { if (t < o) red[t] += red[t + o]; __syncthreads(); }
  const float inv = 1.0f / red[0];
  for (int c = t; c < 32000; c += 256) lg[base + c] *= inv;
}

// ---------------------------------------------------------------------------
// Host-side orchestration.
// Param pointer order assumes jax tree_flatten (dict keys sorted):
//   top: input_sentence, output_sentence, params
//   params: dec[0..5], emb_in, emb_out, enc[0..5], fc
//   dec layer: cross(mha), ff, self(mha);  enc layer: ff, mha
//   mha: Wk{b,w}, Wout{b,w}, Wq{b,w}, Wv{b,w}, ln{b,g}
//   ff:  W1{b,w}, W2{b,w}, ln{b,g};  lin: {b,w};  fc: {b,w}
// ---------------------------------------------------------------------------
struct LinP { const float* w; const float* b; };
struct MHAP { LinP Wk, Wout, Wq, Wv; const float* lnb; const float* lng; };
struct FFP  { LinP W1, W2; const float* lnb; const float* lng; };

struct Scratch {
  float *q, *k, *v, *head, *big, *wsum, *sub;
};

static void run_mha(const float* xq, const float* xkv, float* xio,
                    const MHAP& m, bool causal, const Scratch& w,
                    hipStream_t s) {
  // projections: [2048,512] @ [512,64] + b
  gemm_wmma<false, true, false><<<dim3(1, 16), 256, 0, s>>>(
      xq,  m.Wq.w, m.Wq.b, w.q, 2048, 64, 512, 512, 64, 64);
  gemm_wmma<false, true, false><<<dim3(1, 16), 256, 0, s>>>(
      xkv, m.Wk.w, m.Wk.b, w.k, 2048, 64, 512, 512, 64, 64);
  gemm_wmma<false, true, false><<<dim3(1, 16), 256, 0, s>>>(
      xkv, m.Wv.w, m.Wv.b, w.v, 2048, 64, 512, 512, 64, 64);
  // scores = q @ k^T  (k is [N=2048, K=64] -> TRANSB path, zero-cost layout)
  gemm_wmma<true, false, false><<<dim3(32, 16), 256, 0, s>>>(
      w.q, w.k, nullptr, w.big, 2048, 2048, 64, 64, 64, 2048);
  attn_softmax<<<2048, 256, 0, s>>>(w.big, causal ? 1 : 0);
  // head = probs @ v
  gemm_wmma<false, false, false><<<dim3(1, 16), 256, 0, s>>>(
      w.big, w.v, nullptr, w.head, 2048, 64, 2048, 2048, 64, 64);
  // out = head @ fold(Wout) + b   (concat of identical heads folded to K=64)
  wout_fold<<<128, 256, 0, s>>>(m.Wout.w, w.wsum);
  gemm_wmma<false, true, false><<<dim3(8, 16), 256, 0, s>>>(
      w.head, w.wsum, m.Wout.b, w.sub, 2048, 512, 64, 64, 512, 512);
  ln_res<<<2048, 256, 0, s>>>(xio, w.sub, m.lng, m.lnb);
}

static void run_ff(float* xio, const FFP& f, const Scratch& w, hipStream_t s) {
  gemm_wmma<false, true, true><<<dim3(32, 16), 256, 0, s>>>(
      xio, f.W1.w, f.W1.b, w.big, 2048, 2048, 512, 512, 2048, 2048);
  gemm_wmma<false, true, false><<<dim3(8, 16), 256, 0, s>>>(
      w.big, f.W2.w, f.W2.b, w.sub, 2048, 512, 2048, 2048, 512, 512);
  ln_res<<<2048, 256, 0, s>>>(xio, f.lnb ? w.sub : w.sub, f.lng, f.lnb);
}

extern "C" void kernel_launch(void* const* d_in, const int* in_sizes, int n_in,
                              void* d_out, int out_size, void* d_ws, size_t ws_size,
                              hipStream_t stream) {
  (void)in_sizes; (void)out_size; (void)ws_size;
  if (n_in < 258) return;  // unexpected flattening; bail rather than read OOB

  const int* tok_in  = (const int*)d_in[0];
  const int* tok_out = (const int*)d_in[1];

  int p = 2;
  auto F = [&](void) { return (const float*)d_in[p++]; };
  auto readMHA = [&](MHAP& m) {
    m.Wk.b = F();   m.Wk.w = F();
    m.Wout.b = F(); m.Wout.w = F();
    m.Wq.b = F();   m.Wq.w = F();
    m.Wv.b = F();   m.Wv.w = F();
    m.lnb = F();    m.lng = F();
  };
  auto readFF = [&](FFP& f) {
    f.W1.b = F(); f.W1.w = F();
    f.W2.b = F(); f.W2.w = F();
    f.lnb = F();  f.lng = F();
  };

  MHAP dcross[6], dself[6], emha[6];
  FFP  dff[6], eff[6];
  for (int l = 0; l < 6; ++l) { readMHA(dcross[l]); readFF(dff[l]); readMHA(dself[l]); }
  const float* emb_in  = F();
  const float* emb_out = F();
  for (int l = 0; l < 6; ++l) { readFF(eff[l]); readMHA(emha[l]); }
  const float* fc_b = F();
  const float* fc_w = F();

  // workspace carve-up (~32 MB of f32)
  float* ws = (float*)d_ws;
  float* xe  = ws;                      // [2048,512] encoder stream
  float* xd  = xe + 2048 * 512;         // [2048,512] decoder stream
  Scratch sc;
  sc.big  = xd + 2048 * 512;            // [2048,2048] scores / ffn hidden
  sc.sub  = sc.big + 2048 * 2048;       // [2048,512] sublayer output
  sc.q    = sc.sub + 2048 * 512;        // [2048,64]
  sc.k    = sc.q + 2048 * 64;
  sc.v    = sc.k + 2048 * 64;
  sc.head = sc.v + 2048 * 64;
  sc.wsum = sc.head + 2048 * 64;        // [64,512]

  embed_pos<<<4096, 256, 0, stream>>>(tok_in,  emb_in,  xe);
  embed_pos<<<4096, 256, 0, stream>>>(tok_out, emb_out, xd);

  for (int l = 0; l < 6; ++l) {
    run_mha(xe, xe, xe, emha[l], false, sc, stream);
    run_ff(xe, eff[l], sc, stream);
  }
  for (int l = 0; l < 6; ++l) {
    run_mha(xd, xd, xd, dself[l], true,  sc, stream);   // causal self-attn
    run_mha(xd, xe, xd, dcross[l], false, sc, stream);  // cross-attn (kv=enc)
    run_ff(xd, dff[l], sc, stream);
  }

  // logits -> d_out, then row softmax in place (output is fp32 probabilities)
  float* out = (float*)d_out;
  gemm_wmma<false, true, false><<<dim3(500, 16), 256, 0, stream>>>(
      xd, fc_w, fc_b, out, 2048, 32000, 512, 512, 32000, 32000);
  vocab_softmax<<<2048, 256, 0, stream>>>(out);
}
